// GatedIndexedLinear_26345329394133
// MI455X (gfx1250) — compile-verified, matrix operand order checked by reading, not run on hardware
//
#include <hip/hip_runtime.h>
#include <hip/hip_bf16.h>

// CDNA5 / gfx1250: wave32, WMMA 16x16x32 bf16 -> f32 accumulate.

typedef __attribute__((ext_vector_type(8)))  __bf16 v8bf;
typedef __attribute__((ext_vector_type(16))) __bf16 v16bf;
typedef __attribute__((ext_vector_type(8)))  float  v8f;

// Species-batched GEMM:
//   for species s (64 atoms each, contiguous):
//     OUT[(s*64+m), n_off + n, d] = scale * sum_k X[(s*64+m), k, d] * W[s, k, n]
// X is [4096, K, D] (row-major), W is [64, K, N] (row-major),
// OUT is [4096, NOUT, D]. GEMM rows are (atom m, dim d) pairs: 64*D rows per
// species, split into D row-blocks of 64.
//
// Block = 256 threads = 8 wave32s. Block tile = 64 rows x 32 cols.
// Wave w: M-subtile = (w>>1)*16, N-subtile = (w&1)*16 -> one v8f accumulator.
template <int K, int D, int N, int NOUT>
__global__ __launch_bounds__(256) void species_gemm_wmma(
    const float* __restrict__ X, const float* __restrict__ W,
    float* __restrict__ OUT, int n_off, float scale) {
  const int s   = blockIdx.x;   // species (0..63)
  const int rb  = blockIdx.y;   // row block (0..D-1)
  const int nb  = blockIdx.z;   // col block (0..N/32-1)
  const int tid = threadIdx.x;

  // Padded pitch: 40 halves = 80 bytes/row -> every 8-half-aligned element is
  // 16B-aligned (b128 reads) and every 4-half-aligned element is 8B-aligned
  // (b64 staging stores); pitch != multiple of 64B dodges bank conflicts.
  __shared__ __align__(16) __bf16 As[64][40];  // A tile, row-major [row][k]
  __shared__ __align__(16) __bf16 Bt[32][40];  // W tile, transposed [n][k]

  const int wave = tid >> 5;
  const int lane = tid & 31;
  const int half = lane >> 4;   // lane group selects K 8..15 / 24..31
  const int idx  = lane & 15;   // M row (A) / N col (B) within 16x16 tile
  const int mt   = wave >> 1;   // 0..3
  const int nt   = wave & 1;    // 0..1

  const float* Wsp = W + (size_t)s * K * N;

  // B staging coords (fixed per thread): 32x32 tile = 256 quads-of-4-along-k.
  const int bn  = tid & 31;        // n within tile
  const int bkk = (tid >> 5) * 4;  // k quad start

  v8f acc = {};

  for (int k0 = 0; k0 < K; k0 += 32) {
    // ---- Stage A: 64 rows x 32 k, fp32 -> bf16, packed b64 LDS stores.
    // 512 quads / 256 threads = 2 per thread. Consecutive tids cover
    // consecutive k-quads then rows (coalesced; b128 global loads when D==1).
#pragma unroll
    for (int j = 0; j < 2; ++j) {
      const int q   = tid + 256 * j;   // 0..511
      const int row = q >> 3;          // 0..63
      const int kk  = (q & 7) * 4;     // 0,4,...,28
      const int r   = rb * 64 + row;   // species-local (m,d) row
      const int m   = r / D;
      const int d   = r % D;
      const float* src =
          X + (size_t)(s * 64 + m) * K * D + (size_t)(k0 + kk) * D + d;
      float x0, x1, x2, x3;
      if constexpr (D == 1) {
        const float4 f = *(const float4*)src;  // global_load_b128
        x0 = f.x; x1 = f.y; x2 = f.z; x3 = f.w;
      } else {
        x0 = src[0]; x1 = src[D]; x2 = src[2 * D]; x3 = src[3 * D];
      }
      union { __bf16 h[4]; unsigned long long u; } pk;
      pk.h[0] = (__bf16)x0; pk.h[1] = (__bf16)x1;
      pk.h[2] = (__bf16)x2; pk.h[3] = (__bf16)x3;
      *(unsigned long long*)&As[row][kk] = pk.u;  // ds_store_b64
    }
    // ---- Stage W: 32 k x 32 n, stored transposed Bt[n][k]; one quad-of-k per
    // thread -> one b64 LDS store. Global reads coalesced across lanes
    // (consecutive tid -> consecutive n at fixed k).
    {
      const float* src = Wsp + (size_t)(k0 + bkk) * N + nb * 32 + bn;
      const float x0 = src[0];
      const float x1 = src[N];
      const float x2 = src[2 * N];
      const float x3 = src[3 * N];
      union { __bf16 h[4]; unsigned long long u; } pk;
      pk.h[0] = (__bf16)x0; pk.h[1] = (__bf16)x1;
      pk.h[2] = (__bf16)x2; pk.h[3] = (__bf16)x3;
      *(unsigned long long*)&Bt[bn][bkk] = pk.u;  // ds_store_b64
    }
    // Prefetch next K-step's weight tile (dominant HBM traffic) into cache.
    if (k0 + 32 < K) {
      __builtin_prefetch(Wsp + (size_t)(k0 + 32 + bkk) * N + nb * 32 + bn, 0, 0);
    }
    __syncthreads();

    // Gather WMMA fragments per CDNA5 16-bit A 16x32 layout:
    // lane (half,idx): elems 0..7  -> K = half*8 + 0..7
    //                  elems 8..15 -> K = 16 + half*8 + 0..7
    union { v16bf v; v8bf h[2]; } afrag, bfrag;
    const int arow = mt * 16 + idx;
    afrag.h[0] = *(const v8bf*)&As[arow][half * 8];         // ds_load_b128
    afrag.h[1] = *(const v8bf*)&As[arow][16 + half * 8];
    const int bcol = nt * 16 + idx;
    bfrag.h[0] = *(const v8bf*)&Bt[bcol][half * 8];
    bfrag.h[1] = *(const v8bf*)&Bt[bcol][16 + half * 8];

    // D = A x B + C  (f32 accumulate). EXEC is all-1s here (no divergence).
    acc = __builtin_amdgcn_wmma_f32_16x16x32_bf16(
        false, afrag.v, false, bfrag.v, (short)0, acc, false, false);

    __syncthreads();
  }

  // Epilogue: C/D layout — lane(idx) = N col, VGPR v -> M = v + half*8.
  const int n_g = nb * 32 + nt * 16 + idx;
#pragma unroll
  for (int v = 0; v < 8; ++v) {
    const int r = rb * 64 + mt * 16 + v + half * 8;  // species-local row
    const int m = r / D;
    const int d = r % D;
    const size_t o = (size_t)(s * 64 + m) * NOUT * D +
                     (size_t)(n_off + n_g) * D + d;
    OUT[o] = acc[v] * scale;
  }
}

extern "C" void kernel_launch(void* const* d_in, const int* in_sizes, int n_in,
                              void* d_out, int out_size, void* d_ws, size_t ws_size,
                              hipStream_t stream) {
  (void)in_sizes; (void)n_in; (void)out_size; (void)d_ws; (void)ws_size;

  const float* x_0e = (const float*)d_in[0];  // [4096, 256, 1]
  const float* x_1o = (const float*)d_in[1];  // [4096, 128, 3]
  const float* x_2e = (const float*)d_in[2];  // [4096, 128, 5]
  const float* w_s  = (const float*)d_in[3];  // [64, 256*256]
  const float* w_g  = (const float*)d_in[4];  // [64, 256*256]
  const float* w_1o = (const float*)d_in[5];  // [64, 128*128]
  const float* w_2e = (const float*)d_in[6];  // [64, 128*128]
  // d_in[7] = species_counts: uniform 64 per species (B/NUM_IDX) -> s = b>>6.

  float* out_0e = (float*)d_out;                    // [4096, 512, 1]
  float* out_1o = out_0e + (size_t)4096 * 512;      // [4096, 128, 3]
  float* out_2e = out_1o + (size_t)4096 * 128 * 3;  // [4096, 128, 5]

  const float inv_sqrt_256 = 0.0625f;
  const float inv_sqrt_128 = 0.08838834764831845f;

  dim3 blk(256);
  // 0e scalars: cols [0,256) of the 512-wide output
  species_gemm_wmma<256, 1, 256, 512>
      <<<dim3(64, 1, 8), blk, 0, stream>>>(x_0e, w_s, out_0e, 0, inv_sqrt_256);
  // 0e gates: cols [256,512)
  species_gemm_wmma<256, 1, 256, 512>
      <<<dim3(64, 1, 8), blk, 0, stream>>>(x_0e, w_g, out_0e, 256, inv_sqrt_256);
  // 1o: rows = (atom, dim) pairs, D=3
  species_gemm_wmma<128, 3, 128, 128>
      <<<dim3(64, 3, 4), blk, 0, stream>>>(x_1o, w_1o, out_1o, 0, inv_sqrt_128);
  // 2e: D=5
  species_gemm_wmma<128, 5, 128, 128>
      <<<dim3(64, 5, 4), blk, 0, stream>>>(x_2e, w_2e, out_2e, 0, inv_sqrt_128);
}
